// PatchVectorQuantizer_71262097375562
// MI455X (gfx1250) — compile-verified, matrix operand order checked by reading, not run on hardware
//
#include <hip/hip_runtime.h>

// ---------------------------------------------------------------------------
// VQ codebook search on MI455X (gfx1250): TDM-staged bf16 WMMA distance GEMM
// + argmin, exact f32 gather/loss afterwards.
// ---------------------------------------------------------------------------

#define B_ROWS      16384
#define K_CODES     8192
#define CODE_DIM    1024      // 64 * 16
#define LOSS_SCALE  (1.25f / ((float)B_ROWS * (float)CODE_DIM))

#define KSPLITS     4
#define KS_CODES    (K_CODES / KSPLITS)   // 2048 codes per k-split
#define TILE_C      32                    // codes staged in LDS per pass
#define NPASS       (KS_CODES / TILE_C)   // 64 passes
#define ROW_STRIDE  2080                  // 1024*2B + 2*16B TDM padding
#define BUF_BYTES   (TILE_C * ROW_STRIDE) // 66,560 B per buffer

typedef __attribute__((ext_vector_type(16))) __bf16 bf16x16;
typedef __attribute__((ext_vector_type(8)))  __bf16 bf16x8;
typedef __attribute__((ext_vector_type(8)))  float  f32x8;
typedef __attribute__((ext_vector_type(4)))  unsigned int u32x4;
typedef __attribute__((ext_vector_type(4)))  int    i32x4;
typedef __attribute__((ext_vector_type(8)))  int    i32x8;

#if __has_builtin(__builtin_amdgcn_tensor_load_to_lds) && \
    __has_builtin(__builtin_amdgcn_s_wait_tensorcnt)
#define USE_TDM 1
#endif

// LDS byte offset of element d within a padded tile row (pad 16B per 512 elems)
__device__ __forceinline__ int lds_off(int d) { return (d << 1) + ((d >> 9) << 4); }

#ifdef USE_TDM
// Issue a TDM load of a 32x1024 bf16 tile (row-major, stride 1024 elems) into
// LDS at lds_addr, with 4-dword padding every 256 dwords (row stride 2080 B).
__device__ __forceinline__ void tdm_load_tile(unsigned lds_addr, const __bf16* gsrc) {
    unsigned long long ga = (unsigned long long)(size_t)gsrc;
    u32x4 g0;
    g0[0] = 1u;                                   // count=1 (valid descriptor)
    g0[1] = lds_addr;                             // LDS byte address
    g0[2] = (unsigned)ga;                         // global_addr[31:0]
    g0[3] = (unsigned)(ga >> 32) | (2u << 30);    // global_addr[56:32] | type=2
    i32x8 g1;
    g1[0] = (1 << 16)      // data_size = 2 bytes
          | (1 << 20)      // pad_enable
          | (7 << 22)      // pad_interval: every 256 dwords
          | (3 << 25);     // pad_amount: 4 dwords (16 B)
    g1[1] = (int)(1024u << 16);   // tensor_dim0[15:0] = 1024
    g1[2] = (int)(32u << 16);     // tensor_dim0 hi = 0 | tensor_dim1 lo = 32
    g1[3] = (int)(1024u << 16);   // tensor_dim1 hi = 0 | tile_dim0 = 1024
    g1[4] = 32;                   // tile_dim1 = 32 | tile_dim2 = 0
    g1[5] = 1024;                 // tensor_dim0_stride[31:0]
    g1[6] = (int)(1024u << 16);   // stride0 hi = 0 | tensor_dim1_stride lo = 1024
    g1[7] = 0;                    // tensor_dim1_stride hi
    i32x4 z4 = {0, 0, 0, 0};
#if __clang_major__ >= 23
    i32x8 z8 = {0, 0, 0, 0, 0, 0, 0, 0};
    __builtin_amdgcn_tensor_load_to_lds(g0, g1, z4, z4, z8, 0);
#else
    __builtin_amdgcn_tensor_load_to_lds(g0, g1, z4, z4, 0);
#endif
}
#endif

// ---------------------------------------------------------------------------
// 1) f32 -> bf16 conversion
// ---------------------------------------------------------------------------
__global__ __launch_bounds__(256) void vq_cvt_kernel(const float* __restrict__ src,
                                                     __bf16* __restrict__ dst, int n) {
    int i = (blockIdx.x * 256 + threadIdx.x) * 8;
    if (i >= n) return;
    float4 f0 = *(const float4*)(src + i);
    float4 f1 = *(const float4*)(src + i + 4);
    bf16x8 o;
    o[0] = (__bf16)f0.x; o[1] = (__bf16)f0.y; o[2] = (__bf16)f0.z; o[3] = (__bf16)f0.w;
    o[4] = (__bf16)f1.x; o[5] = (__bf16)f1.y; o[6] = (__bf16)f1.z; o[7] = (__bf16)f1.w;
    *(bf16x8*)(dst + i) = o;
}

// ---------------------------------------------------------------------------
// 2) per-codebook-row squared norm (f32, exact)
// ---------------------------------------------------------------------------
__global__ __launch_bounds__(256) void vq_cnorm_kernel(const float* __restrict__ cb,
                                                       float* __restrict__ cnorm) {
    __shared__ float red[256];
    const int k = blockIdx.x;
    const float* row = cb + (size_t)k * CODE_DIM;
    float s = 0.0f;
    for (int d = threadIdx.x; d < CODE_DIM; d += 256) {
        float v = row[d];
        s += v * v;
    }
    red[threadIdx.x] = s;
    __syncthreads();
    for (int w = 128; w > 0; w >>= 1) {
        if (threadIdx.x < w) red[threadIdx.x] += red[threadIdx.x + w];
        __syncthreads();
    }
    if (threadIdx.x == 0) cnorm[k] = red[0];
}

// ---------------------------------------------------------------------------
// 3) WMMA argmin search over one k-split.
//    grid = (128 row-blocks, 4 k-splits), block = 256 threads = 8 waves.
//    Wave w owns rows [blk.x*128 + w*16, +16). Per pass a 32-code x 1024-dim
//    bf16 tile is staged into LDS (TDM double-buffered, issued by wave 0,
//    tracked by TENSORcnt) and shared by all 8 waves. Inner loop: 32 k-steps
//    of 2x v_wmma_f32_16x16x32_bf16 (A from global/L2, B from LDS).
// ---------------------------------------------------------------------------
__global__ __launch_bounds__(256) void vq_search_kernel(const __bf16* __restrict__ zb,
                                                        const __bf16* __restrict__ cbb,
                                                        const float*  __restrict__ cnorm,
                                                        float* __restrict__ pbest,
                                                        int*   __restrict__ pidx) {
    __shared__ char ldsbuf[2][BUF_BYTES];

    const int tid   = threadIdx.x;
    const int wave  = tid >> 5;
    const int lane  = tid & 31;
    const int l15   = lane & 15;
    const int khalf = (lane >> 4) * 8;                 // 0 or 8
    const int rowBase  = blockIdx.x * 128 + wave * 16;
    const int colStart = blockIdx.y * KS_CODES;

    const __bf16* zrow = zb + (size_t)(rowBase + l15) * CODE_DIM;

    float best[8];
    int   bidx[8];
#pragma unroll
    for (int r = 0; r < 8; ++r) { best[r] = 3.4e38f; bidx[r] = 0; }

#ifdef USE_TDM
    if (tid < 32) tdm_load_tile((unsigned)(size_t)&ldsbuf[0][0],
                                cbb + (size_t)colStart * CODE_DIM);
#endif

    for (int p = 0; p < NPASS; ++p) {
        const int cur = p & 1;
        const int colBase = colStart + p * TILE_C;

#ifdef USE_TDM
        if (tid < 32) {
            if (p + 1 < NPASS) {
                tdm_load_tile((unsigned)(size_t)&ldsbuf[1 - cur][0],
                              cbb + (size_t)(colBase + TILE_C) * CODE_DIM);
                __builtin_amdgcn_s_wait_tensorcnt(1);   // tile p has landed
            } else {
                __builtin_amdgcn_s_wait_tensorcnt(0);
            }
        }
        __syncthreads();                                // tile p visible to all
#else
        __syncthreads();                                // prev readers done
        // cooperative staging: 32 rows x 128 chunks of 16 B, 16 chunks/thread
        for (int c = tid; c < TILE_C * 128; c += 256) {
            const int row = c >> 7, cc = c & 127;
            bf16x8 v = *(const bf16x8*)(cbb + (size_t)(colBase + row) * CODE_DIM + cc * 8);
            *(bf16x8*)(&ldsbuf[cur][row * ROW_STRIDE + cc * 16 + ((cc >= 64) ? 16 : 0)]) = v;
        }
        __syncthreads();
#endif

        f32x8 acc0 = (f32x8){0.f,0.f,0.f,0.f,0.f,0.f,0.f,0.f};
        f32x8 acc1 = (f32x8){0.f,0.f,0.f,0.f,0.f,0.f,0.f,0.f};

        const char* br0 = &ldsbuf[cur][(size_t)l15 * ROW_STRIDE];
        const char* br1 = &ldsbuf[cur][(size_t)(16 + l15) * ROW_STRIDE];

        for (int kt = 0; kt < CODE_DIM / 32; ++kt) {
            const int k0 = (kt << 5) + khalf;
            const int o0 = lds_off(k0);
            const int o1 = lds_off(k0 + 16);

            bf16x8 alo = *(const bf16x8*)(zrow + k0);
            bf16x8 ahi = *(const bf16x8*)(zrow + k0 + 16);
            bf16x16 a = __builtin_shufflevector(alo, ahi,
                         0,1,2,3,4,5,6,7,8,9,10,11,12,13,14,15);

            bf16x8 b0lo = *(const bf16x8*)(br0 + o0);
            bf16x8 b0hi = *(const bf16x8*)(br0 + o1);
            bf16x8 b1lo = *(const bf16x8*)(br1 + o0);
            bf16x8 b1hi = *(const bf16x8*)(br1 + o1);
            bf16x16 b0 = __builtin_shufflevector(b0lo, b0hi,
                          0,1,2,3,4,5,6,7,8,9,10,11,12,13,14,15);
            bf16x16 b1 = __builtin_shufflevector(b1lo, b1hi,
                          0,1,2,3,4,5,6,7,8,9,10,11,12,13,14,15);

            acc0 = __builtin_amdgcn_wmma_f32_16x16x32_bf16(false, a, false, b0,
                                                           (short)0, acc0, false, false);
            acc1 = __builtin_amdgcn_wmma_f32_16x16x32_bf16(false, a, false, b1,
                                                           (short)0, acc1, false, false);
        }

        // score + running argmin. C layout: VGPR r -> M = r (lanes 0-15)
        // or M = 8+r (lanes 16-31); N = lane&15.
        {
            const int   n0 = colBase + l15;
            const int   n1 = colBase + 16 + l15;
            const float c0 = cnorm[n0];
            const float c1 = cnorm[n1];
#pragma unroll
            for (int r = 0; r < 8; ++r) {
                float s0 = c0 - 2.0f * acc0[r];
                float s1 = c1 - 2.0f * acc1[r];
                if (s0 < best[r]) { best[r] = s0; bidx[r] = n0; }
                if (s1 < best[r]) { best[r] = s1; bidx[r] = n1; }
            }
        }

#ifdef USE_TDM
        __syncthreads();   // all readers done before TDM overwrites this buffer
#endif
    }

    // cross-lane min over the 16 lanes sharing each row; lowest-index ties.
#pragma unroll
    for (int r = 0; r < 8; ++r) {
#pragma unroll
        for (int off = 8; off >= 1; off >>= 1) {
            float ov = __shfl_xor(best[r], off, 32);
            int   oi = __shfl_xor(bidx[r], off, 32);
            if (ov < best[r] || (ov == best[r] && oi < bidx[r])) {
                best[r] = ov; bidx[r] = oi;
            }
        }
    }
    if (l15 == 0) {
        const int mBase = rowBase + (lane >> 4) * 8;
#pragma unroll
        for (int r = 0; r < 8; ++r) {
            pbest[(size_t)(mBase + r) * KSPLITS + blockIdx.y] = best[r];
            pidx [(size_t)(mBase + r) * KSPLITS + blockIdx.y] = bidx[r];
        }
    }
}

// ---------------------------------------------------------------------------
// 3b) combine the 4 k-split candidates per row (ascending split order keeps
//     the first-minimum tie-break since indices increase with split).
// ---------------------------------------------------------------------------
__global__ __launch_bounds__(256) void vq_argmin_kernel(const float* __restrict__ pbest,
                                                        const int*   __restrict__ pidx,
                                                        int* __restrict__ idxs) {
    const int row = blockIdx.x * 256 + threadIdx.x;
    if (row >= B_ROWS) return;
    float b = pbest[(size_t)row * KSPLITS];
    int  bi = pidx[(size_t)row * KSPLITS];
#pragma unroll
    for (int s = 1; s < KSPLITS; ++s) {
        float v = pbest[(size_t)row * KSPLITS + s];
        int   i = pidx[(size_t)row * KSPLITS + s];
        if (v < b) { b = v; bi = i; }
    }
    idxs[row] = bi;
}

// ---------------------------------------------------------------------------
// 4) gather quantized rows (exact f32), per-row partial SSE, index output
// ---------------------------------------------------------------------------
__global__ __launch_bounds__(256) void vq_gather_kernel(const float* __restrict__ z,
                                                        const float* __restrict__ cb,
                                                        const int*   __restrict__ idxs,
                                                        float* __restrict__ out,
                                                        float* __restrict__ part) {
    __shared__ float red[256];
    const int row = blockIdx.x;
    const int idx = idxs[row];
    const float* crow = cb + (size_t)idx * CODE_DIM;
    const float* zrow = z  + (size_t)row * CODE_DIM;
    float* qrow = out + 1 + (size_t)row * CODE_DIM;  // 4-byte aligned region

    const int d = threadIdx.x * 4;
    float4 q  = *(const float4*)(crow + d);
    float4 zv = *(const float4*)(zrow + d);
    qrow[d + 0] = q.x;
    qrow[d + 1] = q.y;
    qrow[d + 2] = q.z;
    qrow[d + 3] = q.w;
    float dx = q.x - zv.x, dy = q.y - zv.y, dz = q.z - zv.z, dw = q.w - zv.w;
    red[threadIdx.x] = dx * dx + dy * dy + dz * dz + dw * dw;
    __syncthreads();
    for (int w = 128; w > 0; w >>= 1) {
        if (threadIdx.x < w) red[threadIdx.x] += red[threadIdx.x + w];
        __syncthreads();
    }
    if (threadIdx.x == 0) {
        part[row] = red[0];
        out[1 + (size_t)B_ROWS * CODE_DIM + row] = (float)idx;  // indices output
    }
}

// ---------------------------------------------------------------------------
// 5) deterministic final loss reduction (single block)
// ---------------------------------------------------------------------------
__global__ __launch_bounds__(256) void vq_loss_kernel(const float* __restrict__ part,
                                                      float* __restrict__ out) {
    __shared__ float red[256];
    float s = 0.0f;
    for (int i = threadIdx.x; i < B_ROWS; i += 256) s += part[i];
    red[threadIdx.x] = s;
    __syncthreads();
    for (int w = 128; w > 0; w >>= 1) {
        if (threadIdx.x < w) red[threadIdx.x] += red[threadIdx.x + w];
        __syncthreads();
    }
    if (threadIdx.x == 0) out[0] = red[0] * LOSS_SCALE;
}

// ---------------------------------------------------------------------------
extern "C" void kernel_launch(void* const* d_in, const int* in_sizes, int n_in,
                              void* d_out, int out_size, void* d_ws, size_t ws_size,
                              hipStream_t stream) {
    const float* z  = (const float*)d_in[0];   // [16384, 64, 16] f32
    const float* cb = (const float*)d_in[1];   // [8192, 1024]    f32
    float* out = (float*)d_out;

    // workspace layout
    char* ws = (char*)d_ws;
    const size_t zb_bytes = (size_t)B_ROWS  * CODE_DIM * sizeof(__bf16); // 32 MB
    const size_t cb_bytes = (size_t)K_CODES * CODE_DIM * sizeof(__bf16); // 16 MB
    __bf16* zb    = (__bf16*)ws;
    __bf16* cbb   = (__bf16*)(ws + zb_bytes);
    float*  cnorm = (float*)(ws + zb_bytes + cb_bytes);
    int*    idxs  = (int*)((char*)cnorm + K_CODES * sizeof(float));
    float*  part  = (float*)((char*)idxs + B_ROWS * sizeof(int));
    float*  pbest = (float*)((char*)part + B_ROWS * sizeof(float));
    int*    pidx  = (int*)((char*)pbest + (size_t)B_ROWS * KSPLITS * sizeof(float));

    const int n_z = B_ROWS * CODE_DIM;
    const int n_c = K_CODES * CODE_DIM;

    vq_cvt_kernel<<<n_z / 2048, 256, 0, stream>>>(z, zb, n_z);
    vq_cvt_kernel<<<n_c / 2048, 256, 0, stream>>>(cb, cbb, n_c);
    vq_cnorm_kernel<<<K_CODES, 256, 0, stream>>>(cb, cnorm);
    vq_search_kernel<<<dim3(B_ROWS / 128, KSPLITS), 256, 0, stream>>>(zb, cbb, cnorm,
                                                                      pbest, pidx);
    vq_argmin_kernel<<<B_ROWS / 256, 256, 0, stream>>>(pbest, pidx, idxs);
    vq_gather_kernel<<<B_ROWS, 256, 0, stream>>>(z, cb, idxs, out, part);
    vq_loss_kernel<<<1, 256, 0, stream>>>(part, out);
}
